// LocalClusterBlock_14740327760105
// MI455X (gfx1250) — matrix operand order repaired and checked by reading, not compiled, not run on hardware
//
#include <hip/hip_runtime.h>
#include <hip/hip_bf16.h>
#include <math.h>

// ---------------------------------------------------------------------------
// Problem constants (from the reference): N=8, C=256, H=W=128, HD=256,
// FC=8, FS=2, CS=8, EPS=1e-5.
// ---------------------------------------------------------------------------
constexpr int kN    = 8;
constexpr int kC    = 256;
constexpr int kH    = 128;
constexpr int kW    = 128;
constexpr int kHW   = kH * kW;            // 16384
constexpr int kRows = kN * kHW;           // 131072 spatial positions
constexpr int kM    = 256;                // clusters = N*FC*FS*FS
constexpr int kL    = 4096;               // 64*64 positions per cluster
constexpr int kS    = 64;                 // 8*8 centers per cluster
constexpr int kCV   = 32;                 // point/value channel split
constexpr float kEps = 1e-5f;
constexpr double kGNCnt = (double)kC * (double)kHW; // 4,194,304 elems per sample

typedef __attribute__((ext_vector_type(16))) __bf16 bf16x16;
typedef __attribute__((ext_vector_type(8)))  float  f32x8;

// ---------------------------------------------------------------------------
// bf16 WMMA GEMM:  out[M,Nn] = A[M,K](bf16,row-major) * Bt[Nn,K](bf16)^T + bias
// One wave computes a 16x64 tile (4 adjacent 16x16 WMMA tiles along N):
// the A fragment is loaded once per K-step and reused by 4
// v_wmma_f32_16x16x32_bf16 issues -> 10 b128 loads per 4 WMMAs instead of 16,
// and 4 independent accumulators keep the matrix pipe busy.
// Epilogue modes:
//   0: plain fp32 [M][Nn] store (+bias)
//   1: projection store into clustered layout ypos[m][l][64ch] fp32
//   2: exact GELU then bf16 [M][Nn] store (MLP hidden)
// ---------------------------------------------------------------------------
__global__ __launch_bounds__(256) void lc_gemm_wmma(
    const __hip_bfloat16* __restrict__ A,
    const __hip_bfloat16* __restrict__ Bt,
    const float* __restrict__ bias,
    void* __restrict__ outp,
    int M, int Nn, int K, int mode)
{
  const int wid  = blockIdx.x * 8 + (threadIdx.x >> 5);
  const int lane = threadIdx.x & 31;
  const int tilesN4 = Nn >> 6;                  // groups of 64 columns
  const int tilesTotal = (M >> 4) * tilesN4;
  if (wid >= tilesTotal) return;

  const int tm  = wid / tilesN4;
  const int tn4 = wid - tm * tilesN4;
  const int rlane = lane & 15;
  const int hi = (lane >> 4) & 1;               // lane-half selector

  const int aRow = tm * 16 + rlane;
  const __hip_bfloat16* Ar = A + (size_t)aRow * K;
  const __hip_bfloat16* Br[4];
  #pragma unroll
  for (int j = 0; j < 4; ++j)
    Br[j] = Bt + (size_t)(tn4 * 64 + j * 16 + rlane) * K;

  f32x8 acc[4];
  #pragma unroll
  for (int j = 0; j < 4; ++j) acc[j] = (f32x8){0.f,0.f,0.f,0.f,0.f,0.f,0.f,0.f};

  union V { bf16x16 v; uint4 q[2]; };

  for (int k0 = 0; k0 < K; k0 += 32) {
    // A 16x32 fragment: elems 0..7 -> K=ka..ka+7, elems 8..15 -> K=ka+16..+23
    V a;
    const int ka = k0 + hi * 8;
    a.q[0] = *(const uint4*)(Ar + ka);
    a.q[1] = *(const uint4*)(Ar + ka + 16);
    // 4 B 32x16 fragments: elems 0..7 -> K=kb..kb+7, elems 8..15 -> K=kb+16..+23 of column j (rows of Bt)
    const int kb = k0 + hi * 8;
    #pragma unroll
    for (int j = 0; j < 4; ++j) {
      V b;
      b.q[0] = *(const uint4*)(Br[j] + kb);
      b.q[1] = *(const uint4*)(Br[j] + kb + 16);
      acc[j] = __builtin_amdgcn_wmma_f32_16x16x32_bf16(
          /*neg_a=*/false, a.v, /*neg_b=*/false, b.v,
          /*c_mod=*/(short)0, acc[j], /*reuse_a=*/false, /*reuse_b=*/false);
    }
  }

  #pragma unroll
  for (int j = 0; j < 4; ++j) {
    const int o = tn4 * 64 + j * 16 + rlane;
    const float bv = bias ? bias[o] : 0.f;
    #pragma unroll
    for (int r = 0; r < 8; ++r) {
      const int gRow = tm * 16 + r + hi * 8;   // C/D layout: VGPR r -> row r(+8)
      float v = acc[j][r] + bv;
      if (mode == 0) {
        ((float*)outp)[(size_t)gRow * Nn + o] = v;
      } else if (mode == 1) {
        // scatter into clustered position-major layout ypos[m][l][cc]
        const int n   = gRow >> 14, pix = gRow & 16383;
        const int hh  = pix >> 7,  ww  = pix & 127;
        const int f   = o >> 6,    cc  = o & 63;
        const int m   = n * 32 + f * 4 + (hh >> 6) * 2 + (ww >> 6);
        const int l   = ((hh & 63) << 6) + (ww & 63);
        ((float*)outp)[(((size_t)m << 12) + l) * 64 + cc] = v;
      } else {
        // exact GELU (matches jax.nn.gelu approximate=False)
        const float g = 0.5f * v * (1.0f + erff(v * 0.70710678118654752f));
        ((__hip_bfloat16*)outp)[(size_t)gRow * Nn + o] = __float2bfloat16(g);
      }
    }
  }
}

// ---------------------------------------------------------------------------
// Layout / precision conversion kernels
// ---------------------------------------------------------------------------
// x[n,c,h,w] fp32 -> xbf[row][c] bf16 (position-major activation matrix)
__global__ void lc_convert_x(const float* __restrict__ x,
                             __hip_bfloat16* __restrict__ xbf)
{
  const int id = blockIdx.x * 256 + threadIdx.x;
  if (id >= kRows * kC) return;
  const int row = id >> 8, c = id & 255;
  const int n = row >> 14, pix = row & 16383;
  xbf[id] = __float2bfloat16(x[(((size_t)n * kC + c) << 14) + pix]);
}

// weight fp32 -> bf16 [O][K]; trans=1 means src is [K][O]
__global__ void lc_convert_w(const float* __restrict__ src,
                             __hip_bfloat16* __restrict__ dst,
                             int O, int K, int trans)
{
  const int id = blockIdx.x * 256 + threadIdx.x;
  if (id >= O * K) return;
  const int o = id / K, k = id - o * K;
  const float v = trans ? src[(size_t)k * O + o] : src[(size_t)o * K + k];
  dst[id] = __float2bfloat16(v);
}

__global__ void lc_zero_f64(double* p, int n)
{
  const int id = blockIdx.x * 256 + threadIdx.x;
  if (id < n) p[id] = 0.0;
}

// ---------------------------------------------------------------------------
// Clustering kernels (operate on ypos[m][l][64] fp32)
// ---------------------------------------------------------------------------
// centers[m][s][c] = mean over the 8x8 spatial block s of ypos
__global__ void lc_centers(const float* __restrict__ ypos,
                           float* __restrict__ centers)
{
  const int id = blockIdx.x * 256 + threadIdx.x;   // m*4096 + s*64 + c
  if (id >= kM * kS * 64) return;
  const int c = id & 63, s = (id >> 6) & 63, m = id >> 12;
  const int u = s >> 3, v = s & 7;
  float sum = 0.f;
  for (int du = 0; du < 8; ++du)
    for (int dv = 0; dv < 8; ++dv) {
      const int l = ((u * 8 + du) << 6) + (v * 8 + dv);
      sum += ypos[(((size_t)m << 12) + l) * 64 + c];
    }
  centers[id] = sum * (1.f / 64.f);
}

// cpn[m][s][0:32] = l2-normalized point half of each center
__global__ void lc_cpn(const float* __restrict__ centers,
                       float* __restrict__ cpn)
{
  const int id = blockIdx.x * 256 + threadIdx.x;   // m*64 + s
  if (id >= kM * kS) return;
  const float* cp = centers + (size_t)id * 64;
  float ss = 0.f;
  for (int c = 0; c < kCV; ++c) ss += cp[c] * cp[c];
  const float sc = 1.f / fmaxf(sqrtf(ss), 1e-12f);
  float* dst = cpn + (size_t)id * kCV;
  for (int c = 0; c < kCV; ++c) dst[c] = cp[c] * sc;
}

// per position: cosine sim vs 64 centers, sigmoid(alpha*sim+beta), max/argmax
__global__ void lc_sim(const float* __restrict__ ypos,
                       const float* __restrict__ cpn,
                       const float* __restrict__ alpha,
                       const float* __restrict__ beta,
                       float* __restrict__ vals,
                       int* __restrict__ idxb)
{
  const int id = blockIdx.x * 256 + threadIdx.x;   // m*4096 + l
  if (id >= kM * kL) return;
  const int m = id >> 12;
  const float* xp = ypos + (size_t)id * 64;
  float xr[kCV];
  float ss = 0.f;
  #pragma unroll
  for (int c = 0; c < kCV; ++c) { xr[c] = xp[c]; ss += xr[c] * xr[c]; }
  const float sc = 1.f / fmaxf(sqrtf(ss), 1e-12f);
  const float al = alpha[0], be = beta[0];
  const float* cp = cpn + (size_t)m * kS * kCV;
  float best = -3.0e38f;
  int bs = 0;
  for (int s = 0; s < kS; ++s) {
    float d = 0.f;
    #pragma unroll
    for (int c = 0; c < kCV; ++c) d += xr[c] * cp[s * kCV + c];
    d *= sc;
    const float sg = 1.f / (1.f + expf(-(al * d + be)));
    if (sg > best) { best = sg; bs = s; }   // '>' keeps first max, like argmax
  }
  vals[id] = best;
  idxb[id] = bs;
}

// nc[m][s][0:32] = center value half; nc[m][s][32] = 1 (augmented column)
__global__ void lc_nc_init(const float* __restrict__ centers,
                           float* __restrict__ nc)
{
  const int id = blockIdx.x * 256 + threadIdx.x;   // (m*64+s)*33 + c
  if (id >= kM * kS * 33) return;
  const int c = id % 33, ms = id / 33;
  nc[id] = (c < kCV) ? centers[(size_t)ms * 64 + kCV + c] : 1.0f;
}

// scatter: nc[m][idx] += vals * [x_value, 1]
__global__ void lc_scatter(const float* __restrict__ ypos,
                           const float* __restrict__ vals,
                           const int* __restrict__ idxb,
                           float* __restrict__ nc)
{
  const int id = blockIdx.x * 256 + threadIdx.x;   // m*4096 + l
  if (id >= kM * kL) return;
  const int m = id >> 12;
  const int s = idxb[id];
  const float v = vals[id];
  const float* xv = ypos + (size_t)id * 64 + kCV;  // value half
  float* ncp = nc + ((size_t)m * kS + s) * 33;
  #pragma unroll 4
  for (int c = 0; c < kCV; ++c) atomicAdd(&ncp[c], v * xv[c]);
  atomicAdd(&ncp[32], v);
}

// gather: nx[row][f*32+cc] = vals * (nc[m][idx][cc] / nc[m][idx][32]) -> bf16
__global__ void lc_gather(const float* __restrict__ vals,
                          const int* __restrict__ idxb,
                          const float* __restrict__ nc,
                          __hip_bfloat16* __restrict__ nxbf)
{
  const int id = blockIdx.x * 256 + threadIdx.x;   // row*256 + op
  if (id >= kRows * kC) return;
  const int row = id >> 8, op = id & 255;
  const int n = row >> 14, pix = row & 16383;
  const int hh = pix >> 7, ww = pix & 127;
  const int f = op >> 5, cc = op & 31;
  const int m = n * 32 + f * 4 + (hh >> 6) * 2 + (ww >> 6);
  const int l = ((hh & 63) << 6) + (ww & 63);
  const int ml = (m << 12) + l;
  const int s = idxb[ml];
  const float v = vals[ml];
  const float* ncp = nc + ((size_t)m * kS + s) * 33;
  nxbf[id] = __float2bfloat16(v * (ncp[cc] / ncp[32]));
}

// ---------------------------------------------------------------------------
// GroupNorm (1 group == per-sample) stats: LDS tree reduction + f64 atomics
// src is position-major [131072][256]; 256 consecutive rows share one sample
// ---------------------------------------------------------------------------
__global__ __launch_bounds__(256) void lc_gn_stats(const float* __restrict__ src,
                                                   double* __restrict__ stats)
{
  __shared__ double ssum[256];
  __shared__ double sssq[256];
  const int row = blockIdx.x * 256 + threadIdx.x;
  const float* p = src + (size_t)row * kC;
  float s = 0.f, q = 0.f;
  for (int c = 0; c < kC; ++c) { const float v = p[c]; s += v; q += v * v; }
  ssum[threadIdx.x] = (double)s;
  sssq[threadIdx.x] = (double)q;
  __syncthreads();
  for (int off = 128; off > 0; off >>= 1) {
    if (threadIdx.x < off) {
      ssum[threadIdx.x] += ssum[threadIdx.x + off];
      sssq[threadIdx.x] += sssq[threadIdx.x + off];
    }
    __syncthreads();
  }
  if (threadIdx.x == 0) {
    const int n = (blockIdx.x * 256) >> 14;
    atomicAdd(&stats[n * 2 + 0], ssum[0]);
    atomicAdd(&stats[n * 2 + 1], sssq[0]);
  }
}

// apply GN0 to new_x and build concat MLP input tbf[row][512] bf16
__global__ void lc_gn0_concat(const __hip_bfloat16* __restrict__ xbf,
                              const float* __restrict__ newx,
                              const double* __restrict__ stats,
                              const float* __restrict__ gw,
                              const float* __restrict__ gb,
                              __hip_bfloat16* __restrict__ tbf)
{
  const int id = blockIdx.x * 256 + threadIdx.x;   // row*512 + c
  if (id >= kRows * 512) return;
  const int row = id >> 9, c = id & 511;
  if (c < kC) {
    tbf[id] = xbf[(size_t)row * kC + c];
  } else {
    const int cc = c - kC;
    const int n = row >> 14;
    const float mu  = (float)(stats[n * 2 + 0] / kGNCnt);
    const float ex2 = (float)(stats[n * 2 + 1] / kGNCnt);
    const float rs  = rsqrtf(ex2 - mu * mu + kEps);
    const float v = (newx[(size_t)row * kC + cc] - mu) * rs * gw[cc] + gb[cc];
    tbf[id] = __float2bfloat16(v);
  }
}

// final: GN1(t2) * layer_scale + x, written back to NCHW fp32
__global__ void lc_final(const float* __restrict__ t2,
                         const double* __restrict__ stats,
                         const float* __restrict__ gw,
                         const float* __restrict__ gb,
                         const float* __restrict__ ls,
                         const float* __restrict__ x,
                         float* __restrict__ out)
{
  const int id = blockIdx.x * 256 + threadIdx.x;   // row*256 + c
  if (id >= kRows * kC) return;
  const int row = id >> 8, c = id & 255;
  const int n = row >> 14, pix = row & 16383;
  const float mu  = (float)(stats[n * 2 + 0] / kGNCnt);
  const float ex2 = (float)(stats[n * 2 + 1] / kGNCnt);
  const float rs  = rsqrtf(ex2 - mu * mu + kEps);
  const float v = ((t2[id] - mu) * rs * gw[c] + gb[c]) * ls[c];
  const size_t oi = (((size_t)n * kC + c) << 14) + pix;
  out[oi] = v + x[oi];
}

// ---------------------------------------------------------------------------
// Host launcher
// ---------------------------------------------------------------------------
extern "C" void kernel_launch(void* const* d_in, const int* in_sizes, int n_in,
                              void* d_out, int out_size, void* d_ws, size_t ws_size,
                              hipStream_t stream)
{
  (void)in_sizes; (void)n_in; (void)out_size; (void)ws_size;
  const float* x       = (const float*)d_in[0];
  const float* proj_w  = (const float*)d_in[1];
  const float* proj_b  = (const float*)d_in[2];
  const float* merge_w = (const float*)d_in[3];
  const float* merge_b = (const float*)d_in[4];
  const float* alpha   = (const float*)d_in[5];
  const float* beta    = (const float*)d_in[6];
  const float* g0_w    = (const float*)d_in[7];
  const float* g0_b    = (const float*)d_in[8];
  const float* mlp_w0  = (const float*)d_in[9];
  const float* mlp_b0  = (const float*)d_in[10];
  const float* mlp_w1  = (const float*)d_in[11];
  const float* mlp_b1  = (const float*)d_in[12];
  const float* g1_w    = (const float*)d_in[13];
  const float* g1_b    = (const float*)d_in[14];
  const float* lscale  = (const float*)d_in[15];
  float* out = (float*)d_out;

  // Workspace carve-up (region reuse keeps total ~660 MB)
  char* p = (char*)d_ws;
  auto carve = [&](size_t bytes) -> void* {
    void* r = (void*)p;
    p += (bytes + 255) & ~(size_t)255;
    return r;
  };
  __hip_bfloat16* xbf     = (__hip_bfloat16*)carve((size_t)kRows * kC * 2);
  __hip_bfloat16* projwbf = (__hip_bfloat16*)carve((size_t)512 * 256 * 2);
  __hip_bfloat16* mergwbf = (__hip_bfloat16*)carve((size_t)256 * 256 * 2);
  __hip_bfloat16* w0bf    = (__hip_bfloat16*)carve((size_t)512 * 512 * 2);
  __hip_bfloat16* w1bf    = (__hip_bfloat16*)carve((size_t)256 * 512 * 2);
  double*         stats0  = (double*)carve(16 * sizeof(double));
  double*         stats1  = (double*)carve(16 * sizeof(double));
  float*          ypos    = (float*)carve((size_t)kM * kL * 64 * 4);   // 268 MB
  float*          centers = (float*)carve((size_t)kM * kS * 64 * 4);
  float*          cpn     = (float*)carve((size_t)kM * kS * kCV * 4);
  float*          vals    = (float*)carve((size_t)kM * kL * 4);
  int*            idxb    = (int*)carve((size_t)kM * kL * 4);
  float*          nc      = (float*)carve((size_t)kM * kS * 33 * 4);
  __hip_bfloat16* nxbf    = (__hip_bfloat16*)carve((size_t)kRows * kC * 2);
  float*          newx    = (float*)carve((size_t)kRows * kC * 4);     // 134 MB
  __hip_bfloat16* tbf     = (__hip_bfloat16*)carve((size_t)kRows * 512 * 2);
  // Region reuse: ypos (268 MB) is dead after the scatter; newx after concat.
  __hip_bfloat16* h1bf = (__hip_bfloat16*)ypos;   // 134 MB <= 268 MB
  float*          t2   = newx;                    // 134 MB == 134 MB

  auto nb = [](long n) { return (int)((n + 255) / 256); };
  // one wave per 16x64 tile, 8 waves per block
  auto gemm_blocks = [](int M, int Nn) { return (((M >> 4) * (Nn >> 6)) + 7) / 8; };

  // 0) zero GN stats
  lc_zero_f64<<<1, 256, 0, stream>>>(stats0, 16);
  lc_zero_f64<<<1, 256, 0, stream>>>(stats1, 16);

  // 1) precision/layout conversions
  lc_convert_x<<<nb((long)kRows * kC), 256, 0, stream>>>(x, xbf);
  lc_convert_w<<<nb(512 * 256), 256, 0, stream>>>(proj_w,  projwbf, 512, 256, 0);
  lc_convert_w<<<nb(256 * 256), 256, 0, stream>>>(merge_w, mergwbf, 256, 256, 0);
  lc_convert_w<<<nb(512 * 512), 256, 0, stream>>>(mlp_w0,  w0bf,    512, 512, 1);
  lc_convert_w<<<nb(256 * 512), 256, 0, stream>>>(mlp_w1,  w1bf,    256, 512, 1);

  // 2) projection GEMM -> clustered layout ypos[m][l][64]
  lc_gemm_wmma<<<gemm_blocks(kRows, 512), 256, 0, stream>>>(
      xbf, projwbf, proj_b, ypos, kRows, 512, 256, 1);

  // 3) clustering
  lc_centers<<<nb((long)kM * kS * 64), 256, 0, stream>>>(ypos, centers);
  lc_cpn<<<nb((long)kM * kS), 256, 0, stream>>>(centers, cpn);
  lc_sim<<<nb((long)kM * kL), 256, 0, stream>>>(ypos, cpn, alpha, beta, vals, idxb);
  lc_nc_init<<<nb((long)kM * kS * 33), 256, 0, stream>>>(centers, nc);
  lc_scatter<<<nb((long)kM * kL), 256, 0, stream>>>(ypos, vals, idxb, nc);
  lc_gather<<<nb((long)kRows * kC), 256, 0, stream>>>(vals, idxb, nc, nxbf);

  // 4) merge GEMM -> new_x fp32 [row][256]
  lc_gemm_wmma<<<gemm_blocks(kRows, 256), 256, 0, stream>>>(
      nxbf, mergwbf, merge_b, newx, kRows, 256, 256, 0);

  // 5) GN0 + concat -> tbf [row][512] bf16
  lc_gn_stats<<<kRows / 256, 256, 0, stream>>>(newx, stats0);
  lc_gn0_concat<<<nb((long)kRows * 512), 256, 0, stream>>>(
      xbf, newx, stats0, g0_w, g0_b, tbf);

  // 6) MLP: 512 -> 512 (GELU, bf16 out), then 512 -> 256 (fp32 out)
  lc_gemm_wmma<<<gemm_blocks(kRows, 512), 256, 0, stream>>>(
      tbf, w0bf, mlp_b0, h1bf, kRows, 512, 512, 2);
  lc_gemm_wmma<<<gemm_blocks(kRows, 256), 256, 0, stream>>>(
      h1bf, w1bf, mlp_b1, t2, kRows, 256, 512, 0);

  // 7) GN1 + layer-scale + residual -> NCHW fp32 output
  lc_gn_stats<<<kRows / 256, 256, 0, stream>>>(t2, stats1);
  lc_final<<<nb((long)kRows * kC), 256, 0, stream>>>(
      t2, stats1, g1_w, g1_b, lscale, x, out);
}
